// TransformerDecoderLayer1D_6141803233619
// MI455X (gfx1250) — compile-verified, matrix operand order checked by reading, not run on hardware
//
#include <hip/hip_runtime.h>

// ---------------------------------------------------------------------------
// Transformer decoder layer for MI455X (gfx1250), wave32 + WMMA bf16.
// GEMMs: v_wmma_f32_16x16x32_bf16, 64x64 per-wave register tile (16 WMMA per
// 16 ds_load_b128), double-buffered LDS fed by global_load_async_to_lds_b128.
// ---------------------------------------------------------------------------

typedef __bf16 bf16_t;
typedef __attribute__((ext_vector_type(16))) __bf16 v16bf;
typedef __attribute__((ext_vector_type(8)))  __bf16 v8bf;
typedef __attribute__((ext_vector_type(8)))  float  v8f;

constexpr int B_  = 16;
constexpr int T_  = 256;
constexpr int C_  = 512;
constexpr int H_  = 8;
constexpr int D_  = 64;      // head dim
constexpr int S_  = 1024;    // src tokens (16*64)
constexpr int F_  = 2048;    // ffn hidden
constexpr int MT  = B_ * T_; // 4096 target rows
constexpr int MS  = B_ * S_; // 16384 src rows

static __device__ __forceinline__ bf16_t f2bf(float f) {
    unsigned u = __builtin_bit_cast(unsigned, f);
    unsigned r = u + 0x7fffu + ((u >> 16) & 1u);     // round-to-nearest-even
    unsigned short h = (unsigned short)(r >> 16);
    return __builtin_bit_cast(bf16_t, h);
}

// ------------------------------- pack kernels ------------------------------

__global__ void k_cvt_bf16(const float* __restrict__ x, bf16_t* __restrict__ y, int n) {
    int i = blockIdx.x * 256 + threadIdx.x;
    if (i < n) y[i] = f2bf(x[i]);
}

// w [K,N] f32 -> wt [N,K] bf16
__global__ void k_pack_wt(const float* __restrict__ w, bf16_t* __restrict__ wt, int K, int N) {
    int i = blockIdx.x * 256 + threadIdx.x;
    if (i < K * N) {
        int k = i / N, n = i % N;
        wt[(long long)n * K + k] = f2bf(w[i]);
    }
}

// src [B,C,H*W] f32 -> [B,H*W,C] bf16
__global__ void k_pack_src(const float* __restrict__ s, bf16_t* __restrict__ d) {
    int i = blockIdx.x * 256 + threadIdx.x;        // over B*C*S
    int b = i / (C_ * S_);
    int r = i % (C_ * S_);
    int c = r / S_;
    int p = r % S_;
    d[((long long)b * S_ + p) * C_ + c] = f2bf(s[i]);
}

// ------------------------------- WMMA GEMM ---------------------------------

constexpr int GEMM_BLOCK = 128;   // 4 waves
constexpr int TILE_M = 256, TILE_N = 64, TILE_K = 32;   // wave tile: 64x64
constexpr int LDS_STRIDE = 40;    // 32 + 8 pad (80B rows, 16B aligned)

enum { EPI_PLAIN = 0, EPI_HEADS = 1, EPI_HEADS_T = 2, EPI_CTX = 3, EPI_SCORES = 4 };

// Async 16B global -> LDS copy (CDNA5 GLOBAL_LOAD_ASYNC_TO_LDS_B128, ASYNCcnt).
static __device__ __forceinline__ void async_cp16(unsigned lds_off, const bf16_t* gptr) {
    asm volatile("global_load_async_to_lds_b128 %0, %1, off"
                 :: "v"(lds_off), "v"(gptr) : "memory");
}

// Load one 16x32 bf16 fragment from LDS tile (ISA 7.12.2 16-bit A layout):
// lanes 0-15: row=lane,    elems 0..7 -> K 0..7,  8..15 -> K 16..23
// lanes16-31: row=lane-16, elems 0..7 -> K 8..15, 8..15 -> K 24..31
static __device__ __forceinline__ v16bf frag_ld(const bf16_t* lds, int row0) {
    int lane = threadIdx.x & 31;
    int r  = row0 + (lane & 15);
    int kb = (lane >> 4) << 3;                       // 0 or 8
    const bf16_t* p = lds + r * LDS_STRIDE;
    v8bf lo = *(const v8bf*)(p + kb);                // ds_load_b128
    v8bf hi = *(const v8bf*)(p + 16 + kb);
    return __builtin_shufflevector(lo, hi, 0,1,2,3,4,5,6,7,8,9,10,11,12,13,14,15);
}

// C[M,N] = A[M,K] * Bt[N,K]^T (+bias)(+mask/scale)(+relu); M mult of 256,
// N mult of 64, K mult of 32. Bt always stored row-major [N,K].
template <int EPI, bool RELU, bool OUT_BF16>
__global__ __launch_bounds__(GEMM_BLOCK)
void k_gemm(const bf16_t* __restrict__ A,  long long sAz, int lda,
            const bf16_t* __restrict__ Bt, long long sBz, int ldb,
            const float*  __restrict__ bias,
            void* __restrict__ outp, long long sOz, int ldc,
            int M, int N, int K, float scale,
            const int* __restrict__ mask, long long maskBz, int maskRs,
            int L, int heads)
{
    __shared__ bf16_t lA[2][TILE_M * LDS_STRIDE];   // 2 x 20 KB
    __shared__ bf16_t lB[2][TILE_N * LDS_STRIDE];   // 2 x  5 KB

    const int tid = threadIdx.x;
    const int z   = blockIdx.z;
    const int m0  = blockIdx.y * TILE_M;
    const int n0  = blockIdx.x * TILE_N;
    const bf16_t* Az = A  + (long long)z * sAz;
    const bf16_t* Bz = Bt + (long long)z * sBz;

    const int wid = tid >> 5;
    const int wm  = wid * 64;                        // wave rows in block tile

    // Staging: chunks of 8 bf16 (16B). A tile = 1024 chunks (8/thread, row
    // step 32); B tile = 256 chunks (2/thread). 10 async b128 copies per
    // thread per K-tile.
    const int rA = tid >> 2, gA = (tid & 3) << 3;
    unsigned laA[2], laB0[2], laB1[2];
    #pragma unroll
    for (int bf = 0; bf < 2; ++bf) {
        laA[bf]  = (unsigned)(size_t)&lA[bf][rA * LDS_STRIDE + gA];
        laB0[bf] = (unsigned)(size_t)&lB[bf][rA * LDS_STRIDE + gA];
        laB1[bf] = (unsigned)(size_t)&lB[bf][(rA + 32) * LDS_STRIDE + gA];
    }
    const bf16_t* pA  = Az + (long long)(m0 + rA) * lda + gA;
    const bf16_t* pB0 = Bz + (long long)(n0 + rA) * ldb + gA;
    const bf16_t* pB1 = Bz + (long long)(n0 + rA + 32) * ldb + gA;

    auto issue = [&](int k0, int bf) {
        #pragma unroll
        for (int j = 0; j < 8; ++j)
            async_cp16(laA[bf] + j * (32 * LDS_STRIDE * 2),
                       pA + (long long)j * 32 * lda + k0);
        async_cp16(laB0[bf], pB0 + k0);
        async_cp16(laB1[bf], pB1 + k0);
    };

    v8f acc[4][4] = {};

    const int nT = K / TILE_K;
    issue(0, 0);                                     // preload tile 0
    for (int t = 0; t < nT; ++t) {
        const int bf = t & 1;
        if (t + 1 < nT) {
            issue((t + 1) * TILE_K, bf ^ 1);         // overlap next tile's copy
            asm volatile("s_wait_asynccnt 0xa" ::: "memory");  // tile t done
        } else {
            asm volatile("s_wait_asynccnt 0x0" ::: "memory");
        }
        __syncthreads();                             // tile t visible block-wide

        v16bf a[4], b[4];
        #pragma unroll
        for (int i = 0; i < 4; ++i) a[i] = frag_ld(lA[bf], wm + i * 16);
        #pragma unroll
        for (int i = 0; i < 4; ++i) b[i] = frag_ld(lB[bf], i * 16);

        #pragma unroll
        for (int mi = 0; mi < 4; ++mi)
            #pragma unroll
            for (int ni = 0; ni < 4; ++ni)
                acc[mi][ni] = __builtin_amdgcn_wmma_f32_16x16x32_bf16(
                    false, a[mi], false, b[ni], (short)0, acc[mi][ni], false, false);

        __syncthreads();                             // done reading buf bf
    }

    // Epilogue. C/D layout: VGPR r, lanes 0-15 -> (M=r, N=lane); 16-31 -> (M=r+8).
    const int lane = tid & 31;
    const int nin  = lane & 15;
    const int mhi  = (lane >> 4) & 1;

    auto store_tile = [&](const v8f& av, int tm, int tn) {
        #pragma unroll
        for (int r = 0; r < 8; ++r) {
            int gm = m0 + tm + mhi * 8 + r;
            int gn = n0 + tn + nin;
            float v = av[r];
            if (bias) v += bias[gn];
            if constexpr (EPI == EPI_SCORES) {
                v *= scale;
                int b = z / heads;
                if (mask[(long long)b * maskBz + (long long)gm * maskRs + gn] == 0)
                    v = -1e9f;
            }
            if constexpr (RELU) v = fmaxf(v, 0.f);
            long long oidx;
            if constexpr (EPI == EPI_PLAIN || EPI == EPI_SCORES) {
                oidx = (long long)z * sOz + (long long)gm * ldc + gn;
            } else if constexpr (EPI == EPI_HEADS) {
                int b = gm / L, l = gm % L, h = gn >> 6, d = gn & 63;
                oidx = (((long long)(b * heads + h) * L + l) << 6) + d;
            } else if constexpr (EPI == EPI_HEADS_T) {
                int b = gm / L, l = gm % L, h = gn >> 6, d = gn & 63;
                oidx = ((long long)(b * heads + h) * 64 + d) * L + l;
            } else { // EPI_CTX: per-(b,h) batch in z, n is head dim
                int b = z / heads, h = z % heads;
                oidx = ((long long)(b * L + gm)) * (heads * 64) + h * 64 + gn;
            }
            if constexpr (OUT_BF16) ((bf16_t*)outp)[oidx] = f2bf(v);
            else                    ((float*) outp)[oidx] = v;
        }
    };
    #pragma unroll
    for (int mi = 0; mi < 4; ++mi)
        #pragma unroll
        for (int ni = 0; ni < 4; ++ni)
            store_tile(acc[mi][ni], wm + mi * 16, ni * 16);
}

// ------------------------------- softmax -----------------------------------
// One row per 256-thread block; len in {256,1024}; writes bf16 probs.
__global__ __launch_bounds__(256)
void k_softmax(const float* __restrict__ S, bf16_t* __restrict__ P, int len) {
    __shared__ float red[8];
    const long long row = blockIdx.x;
    const float* sr = S + row * len;
    bf16_t*      pr = P + row * len;
    const int tid = threadIdx.x;
    const int per = len >> 8;           // 1 or 4
    float vals[4];
    float mx = -1e30f;
    for (int j = 0; j < per; ++j) { vals[j] = sr[tid + (j << 8)]; mx = fmaxf(mx, vals[j]); }
    for (int o = 16; o; o >>= 1) mx = fmaxf(mx, __shfl_xor(mx, o, 32));
    if ((tid & 31) == 0) red[tid >> 5] = mx;
    __syncthreads();
    float m2 = red[0];
    #pragma unroll
    for (int i = 1; i < 8; ++i) m2 = fmaxf(m2, red[i]);
    float sum = 0.f;
    for (int j = 0; j < per; ++j) { vals[j] = __expf(vals[j] - m2); sum += vals[j]; }
    for (int o = 16; o; o >>= 1) sum += __shfl_xor(sum, o, 32);
    __syncthreads();
    if ((tid & 31) == 0) red[tid >> 5] = sum;
    __syncthreads();
    float s2 = 0.f;
    #pragma unroll
    for (int i = 0; i < 8; ++i) s2 += red[i];
    const float inv = 1.f / s2;
    for (int j = 0; j < per; ++j) pr[tid + (j << 8)] = f2bf(vals[j] * inv);
}

// ------------------------------- layernorm ---------------------------------
// One row (C=512) per 128-thread block; optional residual; fp32 + bf16 out.
__global__ __launch_bounds__(128)
void k_layernorm(const float* __restrict__ X, const float* __restrict__ Res,
                 const float* __restrict__ g, const float* __restrict__ b,
                 float* __restrict__ Yf, bf16_t* __restrict__ Yb) {
    __shared__ float red[4];
    const long long row = blockIdx.x;
    const float* xr = X + row * C_;
    const int tid = threadIdx.x;
    float v[4];
    float s = 0.f;
    #pragma unroll
    for (int j = 0; j < 4; ++j) {
        int i = tid + (j << 7);
        float x = xr[i];
        if (Res) x += Res[row * C_ + i];
        v[j] = x; s += x;
    }
    for (int o = 16; o; o >>= 1) s += __shfl_xor(s, o, 32);
    if ((tid & 31) == 0) red[tid >> 5] = s;
    __syncthreads();
    const float mean = (red[0] + red[1] + red[2] + red[3]) * (1.f / C_);
    float vs = 0.f;
    #pragma unroll
    for (int j = 0; j < 4; ++j) { float d = v[j] - mean; vs += d * d; }
    for (int o = 16; o; o >>= 1) vs += __shfl_xor(vs, o, 32);
    __syncthreads();
    if ((tid & 31) == 0) red[tid >> 5] = vs;
    __syncthreads();
    const float var = (red[0] + red[1] + red[2] + red[3]) * (1.f / C_);
    const float inv = rsqrtf(var + 1e-5f);
    #pragma unroll
    for (int j = 0; j < 4; ++j) {
        int i = tid + (j << 7);
        float y = (v[j] - mean) * inv * g[i] + b[i];
        Yf[row * C_ + i] = y;
        if (Yb) Yb[row * C_ + i] = f2bf(y);
    }
}

// ------------------------------- launch ------------------------------------

extern "C" void kernel_launch(void* const* d_in, const int* in_sizes, int n_in,
                              void* d_out, int out_size, void* d_ws, size_t ws_size,
                              hipStream_t stream) {
    const float* tgt      = (const float*)d_in[0];
    const float* src      = (const float*)d_in[1];
    const int*   tgt_mask = (const int*)  d_in[2];
    const int*   src_mask = (const int*)  d_in[3];
    const float* sa_wq = (const float*)d_in[4];  const float* sa_bq = (const float*)d_in[5];
    const float* sa_wk = (const float*)d_in[6];  const float* sa_bk = (const float*)d_in[7];
    const float* sa_wv = (const float*)d_in[8];  const float* sa_bv = (const float*)d_in[9];
    const float* sa_wo = (const float*)d_in[10]; const float* sa_bo = (const float*)d_in[11];
    const float* ln1_g = (const float*)d_in[12]; const float* ln1_b = (const float*)d_in[13];
    const float* ca_wq = (const float*)d_in[14]; const float* ca_bq = (const float*)d_in[15];
    const float* ca_wk = (const float*)d_in[16]; const float* ca_bk = (const float*)d_in[17];
    const float* ca_wv = (const float*)d_in[18]; const float* ca_bv = (const float*)d_in[19];
    const float* ca_wo = (const float*)d_in[20]; const float* ca_bo = (const float*)d_in[21];
    const float* ln2_g = (const float*)d_in[22]; const float* ln2_b = (const float*)d_in[23];
    const float* ff_w1 = (const float*)d_in[24]; const float* ff_b1 = (const float*)d_in[25];
    const float* ff_w2 = (const float*)d_in[26]; const float* ff_b2 = (const float*)d_in[27];
    const float* ln3_g = (const float*)d_in[28]; const float* ln3_b = (const float*)d_in[29];

    // ---- workspace carve (needs ~148 MB) ----
    char* ws = (char*)d_ws;
    size_t off = 0;
    auto carve = [&](size_t bytes) { size_t o = off; off += (bytes + 255) & ~size_t(255); return o; };
    bf16_t* wq1t = (bf16_t*)(ws + carve((size_t)C_ * C_ * 2));
    bf16_t* wk1t = (bf16_t*)(ws + carve((size_t)C_ * C_ * 2));
    bf16_t* wv1t = (bf16_t*)(ws + carve((size_t)C_ * C_ * 2));
    bf16_t* wo1t = (bf16_t*)(ws + carve((size_t)C_ * C_ * 2));
    bf16_t* wq2t = (bf16_t*)(ws + carve((size_t)C_ * C_ * 2));
    bf16_t* wk2t = (bf16_t*)(ws + carve((size_t)C_ * C_ * 2));
    bf16_t* wv2t = (bf16_t*)(ws + carve((size_t)C_ * C_ * 2));
    bf16_t* wo2t = (bf16_t*)(ws + carve((size_t)C_ * C_ * 2));
    bf16_t* w1t  = (bf16_t*)(ws + carve((size_t)C_ * F_ * 2));
    bf16_t* w2t  = (bf16_t*)(ws + carve((size_t)F_ * C_ * 2));
    bf16_t* tgtb = (bf16_t*)(ws + carve((size_t)MT * C_ * 2));
    bf16_t* srcb = (bf16_t*)(ws + carve((size_t)MS * C_ * 2));
    bf16_t* qb   = (bf16_t*)(ws + carve((size_t)B_ * H_ * T_ * D_ * 2));
    bf16_t* kb   = (bf16_t*)(ws + carve((size_t)B_ * H_ * S_ * D_ * 2));
    bf16_t* vtb  = (bf16_t*)(ws + carve((size_t)B_ * H_ * S_ * D_ * 2));
    size_t o_sc = carve((size_t)32 * T_ * S_ * 4);   // 32 MB scores scratch
    size_t o_pb = carve((size_t)32 * T_ * S_ * 2);   // 16 MB probs scratch
    float*  sc   = (float*) (ws + o_sc);
    bf16_t* pb   = (bf16_t*)(ws + o_pb);
    bf16_t* ctxb = (bf16_t*)(ws + carve((size_t)MT * C_ * 2));
    float*  attn = (float*) (ws + carve((size_t)MT * C_ * 4));
    float*  out1 = (float*) (ws + carve((size_t)MT * C_ * 4));
    bf16_t* out1b= (bf16_t*)(ws + carve((size_t)MT * C_ * 2));
    float*  out2 = (float*) (ws + carve((size_t)MT * C_ * 4));
    bf16_t* out2b= (bf16_t*)(ws + carve((size_t)MT * C_ * 2));
    bf16_t* h1b  = (bf16_t*)(ws + o_sc);   // reuse scores region (16 MB <= 32 MB)
    float*  yf   = (float*) (ws + o_pb);   // reuse probs region  ( 8 MB <= 16 MB)

    const dim3 blk(GEMM_BLOCK);

    // ---- pack weights / activations to bf16 ----
    int nw = C_ * C_;
    k_pack_wt<<<(nw + 255) / 256, 256, 0, stream>>>(sa_wq, wq1t, C_, C_);
    k_pack_wt<<<(nw + 255) / 256, 256, 0, stream>>>(sa_wk, wk1t, C_, C_);
    k_pack_wt<<<(nw + 255) / 256, 256, 0, stream>>>(sa_wv, wv1t, C_, C_);
    k_pack_wt<<<(nw + 255) / 256, 256, 0, stream>>>(sa_wo, wo1t, C_, C_);
    k_pack_wt<<<(nw + 255) / 256, 256, 0, stream>>>(ca_wq, wq2t, C_, C_);
    k_pack_wt<<<(nw + 255) / 256, 256, 0, stream>>>(ca_wk, wk2t, C_, C_);
    k_pack_wt<<<(nw + 255) / 256, 256, 0, stream>>>(ca_wv, wv2t, C_, C_);
    k_pack_wt<<<(nw + 255) / 256, 256, 0, stream>>>(ca_wo, wo2t, C_, C_);
    k_pack_wt<<<(C_ * F_ + 255) / 256, 256, 0, stream>>>(ff_w1, w1t, C_, F_);
    k_pack_wt<<<(F_ * C_ + 255) / 256, 256, 0, stream>>>(ff_w2, w2t, F_, C_);
    k_cvt_bf16<<<(MT * C_ + 255) / 256, 256, 0, stream>>>(tgt, tgtb, MT * C_);
    k_pack_src<<<(B_ * C_ * S_ + 255) / 256, 256, 0, stream>>>(src, srcb);

    // ---- self-attention ----
    dim3 gP(C_ / TILE_N, MT / TILE_M, 1);
    k_gemm<EPI_HEADS,   false, true><<<gP, blk, 0, stream>>>(tgtb, 0, C_, wq1t, 0, C_, sa_bq, qb,  0, 0, MT, C_, C_, 1.f, nullptr, 0, 0, T_, H_);
    k_gemm<EPI_HEADS,   false, true><<<gP, blk, 0, stream>>>(tgtb, 0, C_, wk1t, 0, C_, sa_bk, kb,  0, 0, MT, C_, C_, 1.f, nullptr, 0, 0, T_, H_);
    k_gemm<EPI_HEADS_T, false, true><<<gP, blk, 0, stream>>>(tgtb, 0, C_, wv1t, 0, C_, sa_bv, vtb, 0, 0, MT, C_, C_, 1.f, nullptr, 0, 0, T_, H_);

    dim3 gS1(T_ / TILE_N, T_ / TILE_M, B_ * H_);
    k_gemm<EPI_SCORES, false, false><<<gS1, blk, 0, stream>>>(
        qb, (long long)T_ * D_, D_, kb, (long long)T_ * D_, D_, nullptr,
        sc, (long long)T_ * T_, T_, T_, T_, D_, 0.125f,
        tgt_mask, (long long)T_ * T_, T_, T_, H_);
    k_softmax<<<B_ * H_ * T_, 256, 0, stream>>>(sc, pb, T_);
    dim3 gC1(1, T_ / TILE_M, B_ * H_);
    k_gemm<EPI_CTX, false, true><<<gC1, blk, 0, stream>>>(
        pb, (long long)T_ * T_, T_, vtb, (long long)D_ * T_, T_, nullptr,
        ctxb, 0, 0, T_, D_, T_, 1.f, nullptr, 0, 0, T_, H_);
    k_gemm<EPI_PLAIN, false, false><<<gP, blk, 0, stream>>>(ctxb, 0, C_, wo1t, 0, C_, sa_bo, attn, 0, C_, MT, C_, C_, 1.f, nullptr, 0, 0, T_, H_);
    k_layernorm<<<MT, 128, 0, stream>>>(attn, nullptr, ln1_g, ln1_b, out1, out1b);

    // ---- cross-attention ----
    k_gemm<EPI_HEADS, false, true><<<gP, blk, 0, stream>>>(out1b, 0, C_, wq2t, 0, C_, ca_bq, qb, 0, 0, MT, C_, C_, 1.f, nullptr, 0, 0, T_, H_);
    dim3 gPS(C_ / TILE_N, MS / TILE_M, 1);
    k_gemm<EPI_HEADS,   false, true><<<gPS, blk, 0, stream>>>(srcb, 0, C_, wk2t, 0, C_, ca_bk, kb,  0, 0, MS, C_, C_, 1.f, nullptr, 0, 0, S_, H_);
    k_gemm<EPI_HEADS_T, false, true><<<gPS, blk, 0, stream>>>(srcb, 0, C_, wv2t, 0, C_, ca_bv, vtb, 0, 0, MS, C_, C_, 1.f, nullptr, 0, 0, S_, H_);

    for (int c = 0; c < 4; ++c) {           // 32 (b,h) batches per chunk
        int zb = c * 32;
        dim3 gS2(S_ / TILE_N, T_ / TILE_M, 32);
        k_gemm<EPI_SCORES, false, false><<<gS2, blk, 0, stream>>>(
            qb + (long long)zb * T_ * D_, (long long)T_ * D_, D_,
            kb + (long long)zb * S_ * D_, (long long)S_ * D_, D_, nullptr,
            sc, (long long)T_ * S_, S_, T_, S_, D_, 0.125f,
            src_mask + (long long)(zb / H_) * S_, S_, 0, T_, H_);
        k_softmax<<<32 * T_, 256, 0, stream>>>(sc, pb, S_);
        dim3 gC2(1, T_ / TILE_M, 32);
        k_gemm<EPI_CTX, false, true><<<gC2, blk, 0, stream>>>(
            pb, (long long)T_ * S_, S_,
            vtb + (long long)zb * D_ * S_, (long long)D_ * S_, S_, nullptr,
            ctxb + (long long)(zb / H_) * T_ * C_, 0, 0, T_, D_, S_, 1.f, nullptr, 0, 0, T_, H_);
    }
    k_gemm<EPI_PLAIN, false, false><<<gP, blk, 0, stream>>>(ctxb, 0, C_, wo2t, 0, C_, ca_bo, attn, 0, C_, MT, C_, C_, 1.f, nullptr, 0, 0, T_, H_);
    k_layernorm<<<MT, 128, 0, stream>>>(attn, out1, ln2_g, ln2_b, out2, out2b);

    // ---- feedforward ----
    dim3 gF1(F_ / TILE_N, MT / TILE_M, 1);
    k_gemm<EPI_PLAIN, true, true><<<gF1, blk, 0, stream>>>(out2b, 0, C_, w1t, 0, C_, ff_b1, h1b, 0, F_, MT, F_, C_, 1.f, nullptr, 0, 0, T_, H_);
    dim3 gF2(C_ / TILE_N, MT / TILE_M, 1);
    k_gemm<EPI_PLAIN, false, false><<<gF2, blk, 0, stream>>>(h1b, 0, F_, w2t, 0, F_, ff_b2, yf, 0, C_, MT, C_, F_, 1.f, nullptr, 0, 0, T_, H_);
    k_layernorm<<<MT, 128, 0, stream>>>(yf, out2, ln3_g, ln3_b, (float*)d_out, nullptr);
}
